// DomainSPDBatchNorm_73443940761626
// MI455X (gfx1250) — matrix operand order repaired and checked by reading, not compiled, not run on hardware
//
#include <hip/hip_runtime.h>
#include <math.h>

// ---------------------------------------------------------------------------
// Domain-wise SPD BatchNorm on MI455X (gfx1250).
// Matrix functions are GEMM-rich iterations on v_wmma_f32_16x16x32_bf16.
// One 256-thread workgroup (8 waves) per 64x64 matrix.
// f32 master matrices in LDS; per-GEMM bf16 staging (A row-major, B transposed)
// so fragment loads are contiguous ds_load_b128 and each element is converted
// to bf16 exactly once per GEMM.
// ---------------------------------------------------------------------------

#define NDIM   64
#define NELEM  (NDIM * NDIM)         // 4096 floats per matrix
#define NDOM   8
#define LDP    66                    // f32 pitch: even (8B pairs), no bank conflicts
#define BUF    (NDIM * LDP)          // 4224 floats per f32 LDS matrix buffer
#define NBUF   6
#define APITCH 72                    // bf16 staging pitch (36-bank stride, conflict-free)
#define STAGE_BF16 (NDIM * APITCH)   // 4608 bf16 = 2304 floats per staging tile
#define SMEM_FLOATS (NBUF * BUF + 8 + STAGE_BF16 /*A*/ + STAGE_BF16 /*Bt*/)
#define SMEM_BYTES  (SMEM_FLOATS * 4)

#define NS_ITERS 16                  // Newton-Schulz iterations (sqrt/invsqrt)
#define S_LOG    5                   // square roots in inverse scaling-squaring
#define S_EXP    6                   // squarings in scaling-squaring expm

#define ETA  1.0f
#define EPS_ 1e-5f

// workspace layout (float offsets)
#define WS_CNT    0                      // 8
#define WS_SUMSQ  8                      // 8
#define WS_S      16                     // 8
#define WS_BM     32                     // 8*4096
#define WS_BMSQ   (WS_BM    + NDOM*NELEM)
#define WS_BMISQ  (WS_BMSQ  + NDOM*NELEM)   // later reused for rm_isq
#define WS_GT     (WS_BMISQ + NDOM*NELEM)
#define WS_RM     (WS_GT    + NDOM*NELEM)
#define WS_BSQ    (WS_RM    + NDOM*NELEM)   // 4096 (sqrtm(mean))
#define WS_END    (WS_BSQ + NELEM)

typedef __attribute__((ext_vector_type(16))) __bf16 v16bf;
typedef __attribute__((ext_vector_type(8)))  __bf16 v8bf;
typedef __attribute__((ext_vector_type(2)))  __bf16 bf2;
typedef __attribute__((ext_vector_type(8)))  float  v8f;
typedef __attribute__((ext_vector_type(2)))  float  f32x2;

typedef __attribute__((address_space(3))) float  LDSf;
typedef __attribute__((address_space(3))) f32x2  LDSf2;
typedef __attribute__((address_space(3))) __bf16 LDSh;
typedef __attribute__((address_space(3))) v8bf   LDSv8bf;
typedef __attribute__((address_space(3))) bf2    LDSbf2;

// Generic->LDS pointer: low 32 bits of a generic LDS pointer are the DS offset.
__device__ inline LDSf* lds_cast(void* p) {
    return (LDSf*)(unsigned int)(unsigned long long)p;
}

// ---------------------------------------------------------------------------
// 64x64x64 GEMM in LDS: D = A * B   (D distinct from A and B).
// Phase 1: stage A -> bf16 row-major (sA), B -> bf16 transposed (sB, [n][k]).
// Phase 2: 8 waves, two per output column; per wave 2 row-tiles sharing the
//          B fragment. All fragment loads are contiguous ds_load_b128.
// ---------------------------------------------------------------------------
__device__ inline void wg_mm(LDSf* D, const LDSf* A, const LDSf* B,
                             LDSh* sA, LDSh* sB) {
    const int tid = threadIdx.x;
    __syncthreads();
    // ---- stage A (row-major bf16) ----
#pragma unroll
    for (int i = 2 * tid; i < NELEM; i += 512) {
        int r = i >> 6, c = i & 63;
        f32x2 p = *(const LDSf2*)(A + r * LDP + c);
        bf2 q; q.x = (__bf16)p.x; q.y = (__bf16)p.y;
        *(LDSbf2*)(sA + r * APITCH + c) = q;
    }
    // ---- stage B transposed (sB[n][k] = B[k][n], bf16) ----
#pragma unroll
    for (int j = tid; j < 2048; j += 256) {
        int c = j >> 5, r0 = (j & 31) << 1;           // Bt row c, cols r0,r0+1
        float x0 = B[r0 * LDP + c];
        float x1 = B[(r0 + 1) * LDP + c];
        bf2 q; q.x = (__bf16)x0; q.y = (__bf16)x1;
        *(LDSbf2*)(sB + c * APITCH + r0) = q;
    }
    __syncthreads();
    // ---- fragment phase ----
    const int wave = tid >> 5;
    const int lane = tid & 31;
    const int half = lane >> 4;       // 0: lanes 0-15, 1: lanes 16-31
    const int lr   = lane & 15;
    const int tj   = wave >> 1;       // output column tile 0..3
    const int ti0  = wave & 1;        // row tiles ti0 and ti0+2
    const int n    = tj * 16 + lr;
    v8f acc0 = {}, acc1 = {};
    const LDSh* Arow0 = sA + (ti0 * 16 + lr) * APITCH;
    const LDSh* Arow1 = sA + ((ti0 + 2) * 16 + lr) * APITCH;
    const LDSh* Brow  = sB + n * APITCH;
#pragma unroll
    for (int kb = 0; kb < 2; ++kb) {
        const int k0 = kb * 32;
        // B fragment: bf[0..7] = K k0+half*8.., bf[8..15] = K k0+16+half*8..
        v8bf blo = *(const LDSv8bf*)(Brow + k0 + half * 8);
        v8bf bhi = *(const LDSv8bf*)(Brow + k0 + 16 + half * 8);
        v16bf bfr = __builtin_shufflevector(blo, bhi,
            0, 1, 2, 3, 4, 5, 6, 7, 8, 9, 10, 11, 12, 13, 14, 15);
        // A fragments: af[0..7] = K k0+half*8.., af[8..15] = K k0+16+half*8..
        v8bf a0lo = *(const LDSv8bf*)(Arow0 + k0 + half * 8);
        v8bf a0hi = *(const LDSv8bf*)(Arow0 + k0 + 16 + half * 8);
        v8bf a1lo = *(const LDSv8bf*)(Arow1 + k0 + half * 8);
        v8bf a1hi = *(const LDSv8bf*)(Arow1 + k0 + 16 + half * 8);
        v16bf af0 = __builtin_shufflevector(a0lo, a0hi,
            0, 1, 2, 3, 4, 5, 6, 7, 8, 9, 10, 11, 12, 13, 14, 15);
        v16bf af1 = __builtin_shufflevector(a1lo, a1hi,
            0, 1, 2, 3, 4, 5, 6, 7, 8, 9, 10, 11, 12, 13, 14, 15);
        acc0 = __builtin_amdgcn_wmma_f32_16x16x32_bf16(
            false, af0, false, bfr, (short)0, acc0, false, false);
        acc1 = __builtin_amdgcn_wmma_f32_16x16x32_bf16(
            false, af1, false, bfr, (short)0, acc1, false, false);
    }
    // C/D layout: lane lr = n; VGPR r -> row r + half*8
#pragma unroll
    for (int r = 0; r < 8; ++r) {
        D[(ti0 * 16 + half * 8 + r) * LDP + n]       = acc0[r];
        D[((ti0 + 2) * 16 + half * 8 + r) * LDP + n] = acc1[r];
    }
    __syncthreads();
}

// ---------------------------------------------------------------------------
// elementwise helpers on pitched LDS matrices
// ---------------------------------------------------------------------------
__device__ inline void wg_load(LDSf* dst, const float* __restrict__ g) {
    for (int i = threadIdx.x; i < NELEM; i += blockDim.x)
        dst[(i >> 6) * LDP + (i & 63)] = g[i];
    __syncthreads();
}
__device__ inline void wg_store(float* __restrict__ g, const LDSf* src) {
    for (int i = threadIdx.x; i < NELEM; i += blockDim.x)
        g[i] = src[(i >> 6) * LDP + (i & 63)];
    __syncthreads();
}
__device__ inline void wg_atomic_accum(float* g, const LDSf* src) {
    for (int i = threadIdx.x; i < NELEM; i += blockDim.x)
        atomicAdd(&g[i], src[(i >> 6) * LDP + (i & 63)]);
    __syncthreads();
}
__device__ inline void wg_copy(LDSf* dst, const LDSf* src) {
    for (int i = threadIdx.x; i < NELEM; i += blockDim.x) {
        int r = i >> 6, c = i & 63;
        dst[r * LDP + c] = src[r * LDP + c];
    }
    __syncthreads();
}
// dst = a*A + b*B + c*I   (aliasing of dst with A/B is fine; elementwise)
__device__ inline void wg_comb(LDSf* dst, float a, const LDSf* A,
                               float b, const LDSf* B, float c) {
    for (int i = threadIdx.x; i < NELEM; i += blockDim.x) {
        int r = i >> 6, cc = i & 63;
        float v = a * A[r * LDP + cc] + b * B[r * LDP + cc];
        if (r == cc) v += c;
        dst[r * LDP + cc] = v;
    }
    __syncthreads();
}
__device__ inline void wg_scale(LDSf* M, float a) {
    for (int i = threadIdx.x; i < NELEM; i += blockDim.x) {
        int r = i >> 6, c = i & 63;
        M[r * LDP + c] *= a;
    }
    __syncthreads();
}
__device__ inline float wg_trace(const LDSf* A, LDSf* sc) {
    if (threadIdx.x == 0) *sc = 0.f;
    __syncthreads();
    if (threadIdx.x < NDIM)
        atomicAdd((float*)sc, (float)A[threadIdx.x * LDP + threadIdx.x]);
    __syncthreads();
    float v = *sc;
    __syncthreads();
    return v;
}
__device__ inline float wg_dot(const LDSf* A, const LDSf* B, LDSf* sc) {
    if (threadIdx.x == 0) *sc = 0.f;
    __syncthreads();
    float p = 0.f;
    for (int i = threadIdx.x; i < NELEM; i += blockDim.x) {
        int r = i >> 6, c = i & 63;
        p += A[r * LDP + c] * B[r * LDP + c];
    }
    atomicAdd((float*)sc, p);
    __syncthreads();
    float v = *sc;
    __syncthreads();
    return v;
}

// ---------------------------------------------------------------------------
// Coupled Newton-Schulz (Denman-Beavers): y <- sqrtm(A), z <- invsqrtm(A).
// ---------------------------------------------------------------------------
__device__ inline void wg_sqrt_invsqrt(LDSf* y, LDSf* z, const LDSf* A,
                                       LDSf* t, LDSf* w, LDSf* sc,
                                       LDSh* sA, LDSh* sB) {
    float c = wg_trace(A, sc);
    c = fmaxf(c, 1e-20f);
    float inv_c = 1.0f / c;
    LDSf* py = y; LDSf* pz = z; LDSf* pw = w;
    for (int i = threadIdx.x; i < NELEM; i += blockDim.x) {
        int r = i >> 6, cc = i & 63;
        py[r * LDP + cc] = A[r * LDP + cc] * inv_c;
        pz[r * LDP + cc] = (r == cc) ? 1.0f : 0.0f;
    }
    __syncthreads();
    for (int it = 0; it < NS_ITERS; ++it) {
        wg_mm(t, pz, py, sA, sB);               // T = Z*Y
        wg_comb(t, -0.5f, t, 0.f, t, 1.5f);     // M = 1.5 I - 0.5 T
        wg_mm(pw, py, t, sA, sB);               // newY = Y*M
        wg_mm(py, t, pz, sA, sB);               // newZ = M*Z (into old Y slot)
        LDSf* tmp = pz; pz = py; py = pw; pw = tmp;
    }
    float sq = sqrtf(c);
    wg_scale(py, sq);                            // sqrt(A)
    wg_scale(pz, 1.0f / sq);                     // invsqrt(A)
    wg_copy(t, pz);
    if (py != y) wg_copy(y, py);
    wg_copy(z, t);
}

// ---------------------------------------------------------------------------
// dst = logm(m): inverse scaling-and-squaring (S_LOG sqrts + deg-7 Mercator).
// Destroys m. dst,m,y,z,t,w all distinct.
// ---------------------------------------------------------------------------
__device__ inline void wg_logm(LDSf* dst, LDSf* m, LDSf* y, LDSf* z,
                               LDSf* t, LDSf* w, LDSf* sc,
                               LDSh* sA, LDSh* sB) {
    for (int s = 0; s < S_LOG; ++s) {
        wg_sqrt_invsqrt(y, z, m, t, w, sc, sA, sB);
        wg_copy(m, y);
    }
    wg_comb(t, 1.f, m, 0.f, m, -1.f);            // E = m - I
    const float c7 = 1.f / 7.f, c6 = -1.f / 6.f, c5 = 1.f / 5.f,
                c4 = -0.25f, c3 = 1.f / 3.f, c2 = -0.5f, c1 = 1.f;
    wg_comb(y, c7, t, 0.f, t, c6);               // P = c7*E + c6*I
    const float ck[5] = {c5, c4, c3, c2, c1};
#pragma unroll
    for (int k = 0; k < 5; ++k) {
        wg_mm(z, y, t, sA, sB);
        wg_comb(y, 1.f, z, 0.f, z, ck[k]);       // P = P*E + ck*I
    }
    wg_mm(dst, y, t, sA, sB);                    // L = P*E
    wg_scale(dst, (float)(1 << S_LOG));          // *2^S
}

// ---------------------------------------------------------------------------
// dst = expm(a): scaling-and-squaring, deg-8 Taylor Horner. Destroys a.
// dst may alias a; a,y,z distinct from each other.
// ---------------------------------------------------------------------------
__device__ inline void wg_expm(LDSf* dst, LDSf* a, LDSf* y, LDSf* z,
                               LDSh* sA, LDSh* sB) {
    wg_scale(a, 1.0f / (float)(1 << S_EXP));
    wg_comb(y, 1.0f / 8.0f, a, 0.f, a, 1.f);     // P = I + A/8
#pragma unroll
    for (int k = 7; k >= 1; --k) {
        wg_mm(z, a, y, sA, sB);
        wg_comb(y, 1.0f / (float)k, z, 0.f, z, 1.f);   // P = I + (A*P)/k
    }
    LDSf* p = y; LDSf* q = z;
    for (int s = 0; s < S_EXP; ++s) {            // square S_EXP times
        wg_mm(q, p, p, sA, sB);
        LDSf* tmp = p; p = q; q = tmp;
    }
    if (dst != p) wg_copy(dst, p);
}

// smem partition helper
#define SMEM_DECL                                              \
    extern __shared__ float smem_raw[];                        \
    LDSf* smem = lds_cast(smem_raw);                           \
    LDSf* b0 = smem;            LDSf* b1 = smem + BUF;         \
    LDSf* b2 = smem + 2 * BUF;  LDSf* b3 = smem + 3 * BUF;     \
    LDSf* b4 = smem + 4 * BUF;  LDSf* b5 = smem + 5 * BUF;     \
    LDSf* sc = smem + NBUF * BUF;                              \
    LDSh* sA = (LDSh*)(smem + NBUF * BUF + 8);                 \
    LDSh* sB = sA + STAGE_BF16;                                \
    (void)b5;

// ===========================================================================
// Kernels
// ===========================================================================
__global__ void k0_zero(float* ws, int n) {
    int i = blockIdx.x * blockDim.x + threadIdx.x;
    if (i < n) ws[i] = 0.f;
}

__global__ void k1_accum(const float* __restrict__ X,
                         const int* __restrict__ dIdx, float* ws) {
    const int n = blockIdx.x;
    const int dd = dIdx[n];
    const float* Xn = X + (size_t)n * NELEM;
    float* acc = ws + WS_BM + (size_t)dd * NELEM;
    for (int i = threadIdx.x; i < NELEM; i += blockDim.x)
        atomicAdd(&acc[i], Xn[i]);
    if (threadIdx.x == 0) atomicAdd(&ws[WS_CNT + dd], 1.f);
}

__global__ void k2_finalize_mean(float* ws) {
    const int d = blockIdx.x;
    float c = fmaxf(ws[WS_CNT + d], 1.f);
    float inv = 1.0f / c;
    float* bm = ws + WS_BM + (size_t)d * NELEM;
    for (int i = threadIdx.x; i < NELEM; i += blockDim.x) bm[i] *= inv;
}

__global__ void k3_domain_sqrt(float* ws) {
    SMEM_DECL;
    const int d = blockIdx.x;
    wg_load(b0, ws + WS_BM + (size_t)d * NELEM);
    wg_sqrt_invsqrt(b1, b2, b0, b3, b4, sc, sA, sB);
    wg_store(ws + WS_BMSQ  + (size_t)d * NELEM, b1);
    wg_store(ws + WS_BMISQ + (size_t)d * NELEM, b2);
}

__global__ void k4_elem_xt(const float* __restrict__ X,
                           const int* __restrict__ dIdx, float* ws) {
    SMEM_DECL;
    const int n = blockIdx.x;
    const int dd = dIdx[n];
    wg_load(b0, ws + WS_BMISQ + (size_t)dd * NELEM);   // isq
    wg_load(b1, X + (size_t)n * NELEM);                 // X
    wg_mm(b2, b0, b1, sA, sB);                          // isq*X
    wg_mm(b1, b2, b0, sA, sB);                          // S = isq*X*isq
    wg_logm(b2, b1, b0, b3, b4, b5, sc, sA, sB);        // XT
    wg_atomic_accum(ws + WS_GT + (size_t)dd * NELEM, b2);
    float ss = wg_dot(b2, b2, sc);
    if (threadIdx.x == 0) atomicAdd(&ws[WS_SUMSQ + dd], ss);
}

__global__ void k5_domain_update(float* ws, const float* __restrict__ stdp) {
    SMEM_DECL;
    const int d = blockIdx.x;
    const float c = fmaxf(ws[WS_CNT + d], 1.f);
    float* gGT = ws + WS_GT + (size_t)d * NELEM;

    wg_load(b0, gGT);
    wg_scale(b0, 1.0f / c);                             // GT = sum/cnt
    wg_store(gGT, b0);                                  // keep normalized GT
    wg_expm(b1, b0, b2, b3, sA, sB);                    // E = expm(GT)
    wg_load(b0, ws + WS_BMSQ + (size_t)d * NELEM);      // bm_sq
    wg_mm(b2, b0, b1, sA, sB);
    wg_mm(b1, b2, b0, sA, sB);                          // bm_new = sq*E*sq
    wg_logm(b2, b1, b0, b3, b4, b5, sc, sA, sB);        // L = logm(bm_new)
    wg_scale(b2, ETA);
    wg_expm(b1, b2, b0, b3, sA, sB);                    // rm = expm(ETA*L)
    wg_store(ws + WS_RM + (size_t)d * NELEM, b1);
    wg_load(b0, ws + WS_BMISQ + (size_t)d * NELEM);     // bm_isq
    wg_mm(b2, b0, b1, sA, sB);
    wg_mm(b3, b2, b0, sA, sB);                          // C2 = isq*rm*isq
    wg_logm(b1, b3, b0, b2, b4, b5, sc, sA, sB);        // GT2
    wg_load(b0, gGT);                                   // normalized GT
    float dgg2  = wg_dot(b0, b1, sc);
    float dg2g2 = wg_dot(b1, b1, sc);
    float var = ws[WS_SUMSQ + d] / c - 2.f * dgg2 + dg2g2;
    float rv  = (1.f - ETA) * 1.f + ETA * var;
    if (threadIdx.x == 0) ws[WS_S + d] = stdp[0] / sqrtf(rv + EPS_);
    __syncthreads();
    wg_load(b0, ws + WS_RM + (size_t)d * NELEM);        // rm
    wg_sqrt_invsqrt(b1, b2, b0, b3, b4, sc, sA, sB);
    wg_store(ws + WS_BMISQ + (size_t)d * NELEM, b2);    // slot reused: rm_isq
}

__global__ void k5b_sqrt_mean(const float* __restrict__ mean, float* ws) {
    SMEM_DECL;
    wg_load(b0, mean);
    wg_sqrt_invsqrt(b1, b2, b0, b3, b4, sc, sA, sB);    // b1 = sqrtm(mean)
    wg_store(ws + WS_BSQ, b1);
}

__global__ void k6_elem_out(const float* __restrict__ X,
                            const int* __restrict__ dIdx,
                            const float* __restrict__ ws,
                            float* __restrict__ out) {
    SMEM_DECL;
    const int n = blockIdx.x;
    const int dd = dIdx[n];
    const float sv = ws[WS_S + dd];
    wg_load(b0, ws + WS_BMISQ + (size_t)dd * NELEM);    // rm_isq
    wg_load(b1, X + (size_t)n * NELEM);
    wg_mm(b2, b0, b1, sA, sB);
    wg_mm(b1, b2, b0, sA, sB);                          // M = isq*X*isq
    wg_logm(b2, b1, b0, b3, b4, b5, sc, sA, sB);        // inner = logm(M)
    wg_scale(b2, sv);                                   // s * inner
    wg_expm(b1, b2, b0, b3, sA, sB);                    // E = expm(s*inner)
    wg_load(b0, ws + WS_BSQ);                           // B_sq
    wg_mm(b2, b0, b1, sA, sB);
    wg_mm(b1, b2, b0, sA, sB);                          // Xn = B_sq*E*B_sq
    wg_store(out + (size_t)n * NELEM, b1);
}

// ===========================================================================
extern "C" void kernel_launch(void* const* d_in, const int* in_sizes, int n_in,
                              void* d_out, int out_size, void* d_ws,
                              size_t ws_size, hipStream_t stream) {
    (void)n_in; (void)out_size; (void)ws_size;
    const float* X    = (const float*)d_in[0];
    const int*   dIdx = (const int*)d_in[1];
    const float* mean = (const float*)d_in[2];
    const float* stdp = (const float*)d_in[3];
    float* out = (float*)d_out;
    float* ws  = (float*)d_ws;
    const int N = in_sizes[0] / NELEM;   // 4096

    k0_zero<<<(WS_END + 255) / 256, 256, 0, stream>>>(ws, WS_END);
    k1_accum<<<N, 256, 0, stream>>>(X, dIdx, ws);
    k2_finalize_mean<<<NDOM, 256, 0, stream>>>(ws);
    k3_domain_sqrt<<<NDOM, 256, SMEM_BYTES, stream>>>(ws);
    k4_elem_xt<<<N, 256, SMEM_BYTES, stream>>>(X, dIdx, ws);
    k5_domain_update<<<NDOM, 256, SMEM_BYTES, stream>>>(ws, stdp);
    k5b_sqrt_mean<<<1, 256, SMEM_BYTES, stream>>>(mean, ws);
    k6_elem_out<<<N, 256, SMEM_BYTES, stream>>>(X, dIdx, ws, out);
}
